// AttentionBlock_53197464928887
// MI455X (gfx1250) — matrix operand order repaired and checked by reading, not compiled, not run on hardware
//
#include <hip/hip_runtime.h>
#include <hip/hip_bf16.h>

typedef __bf16 bf16_t;
typedef __bf16 v16bf __attribute__((ext_vector_type(16)));
typedef float  v8f   __attribute__((ext_vector_type(8)));

union BFr { v16bf v; uint4 u[2]; };

#define AB_B 4
#define AB_C 64
#define AB_N 4096
#define LOG2E 1.44269504088896340736f

__device__ __forceinline__ v8f wmma_bf16(const BFr& a, const BFr& bb, v8f c)
{
    return __builtin_amdgcn_wmma_f32_16x16x32_bf16(false, a.v, false, bb.v, (short)0, c, false, false);
}

// butterfly max within each 16-lane row on the VALU pipe (v_permlane16_b32)
__device__ __forceinline__ float fmax_bfly16(float v, unsigned s1, unsigned s2)
{
    union { float f; unsigned u; } a, bb;
    a.f  = v;
    bb.u = __builtin_amdgcn_permlane16(a.u, a.u, s1, s2, false, false);
    return fmaxf(v, bb.f);
}

// -------------------- kernel 1: InstanceNorm statistics --------------------
__global__ __launch_bounds__(256) void k_stats(const float* __restrict__ x,
                                               float* __restrict__ mean_out,
                                               float* __restrict__ rstd_out)
{
    const int bc  = blockIdx.x;
    const int tid = threadIdx.x;
    const float* xp = x + ((size_t)bc << 12);
    float s = 0.f, s2 = 0.f;
    for (int i = tid; i < AB_N; i += 256) { float t = xp[i]; s += t; s2 += t * t; }
    #pragma unroll
    for (int off = 16; off; off >>= 1) {
        s  += __shfl_xor(s,  off, 32);
        s2 += __shfl_xor(s2, off, 32);
    }
    __shared__ float rs[8], rs2[8];
    if ((tid & 31) == 0) { rs[tid >> 5] = s; rs2[tid >> 5] = s2; }
    __syncthreads();
    if (tid == 0) {
        float ts = 0.f, ts2 = 0.f;
        for (int i = 0; i < 8; ++i) { ts += rs[i]; ts2 += rs2[i]; }
        float mean = ts * (1.f / AB_N);
        float var  = ts2 * (1.f / AB_N) - mean * mean;   // biased var, as reference
        mean_out[bc] = mean;
        rstd_out[bc] = rsqrtf(var + 1e-5f);
    }
}

// -------------------- kernel 2: fused norm + QKV 1x1 conv (WMMA) --------------------
// Out[o][n] = W[192x64] @ Xn[64x64-tile]; q pre-scaled by 8*log2(e); vt transposed.
__global__ __launch_bounds__(128) void k_qkv(const float* __restrict__ x,
                                             const float* __restrict__ meanp,
                                             const float* __restrict__ rstdp,
                                             const float* __restrict__ w,
                                             const float* __restrict__ bias,
                                             bf16_t* __restrict__ q,
                                             bf16_t* __restrict__ k,
                                             bf16_t* __restrict__ vt)
{
    __shared__ bf16_t sX[64][72];   // [n][c] normalized bf16 (transposed for B-frags)
    const int b    = blockIdx.y;
    const int n0   = blockIdx.x << 6;
    const int tid  = threadIdx.x;
    const int wave = __builtin_amdgcn_readfirstlane(tid >> 5);   // SGPR: scalar branches
    const int lane = tid & 31;
    const int g    = lane >> 4;
    const int ln   = lane & 15;

    // fill sX: thread reads 32 consecutive n of channel c (coalesced), writes transposed
    {
        const int c    = tid >> 1;
        const int nh   = (tid & 1) * 32;
        const float mu = meanp[b * 64 + c];
        const float rs = rstdp[b * 64 + c];
        const float* xr = x + (((size_t)b * 64 + c) << 12) + n0 + nh;
        #pragma unroll
        for (int i = 0; i < 32; ++i)
            sX[nh + i][c] = (bf16_t)((xr[i] - mu) * rs);
    }

    // W A-fragments + bias fragments for this wave's 3 o-tiles
    BFr wa[3][2];
    v8f bfr[3];
    #pragma unroll
    for (int i = 0; i < 3; ++i) {
        const int ot = wave * 3 + i;
        const float* wr = w + (size_t)(ot * 16 + ln) * 64;
        #pragma unroll
        for (int h = 0; h < 2; ++h) {
            #pragma unroll
            for (int e = 0; e < 8; ++e) {
                wa[i][h].v[e]     = (bf16_t)wr[h * 32 + g * 8 + e];
                wa[i][h].v[e + 8] = (bf16_t)wr[h * 32 + 16 + g * 8 + e];
            }
        }
        #pragma unroll
        for (int r = 0; r < 8; ++r) bfr[i][r] = bias[ot * 16 + g * 8 + r];
    }

    __syncthreads();

    #pragma unroll
    for (int j = 0; j < 4; ++j) {
        const bf16_t* xr = &sX[j * 16 + ln][0];
        BFr b0, b1;
        b0.u[0] = *(const uint4*)(xr + g * 8);
        b0.u[1] = *(const uint4*)(xr + 16 + g * 8);
        b1.u[0] = *(const uint4*)(xr + 32 + g * 8);
        b1.u[1] = *(const uint4*)(xr + 48 + g * 8);
        const int n = n0 + j * 16 + ln;
        #pragma unroll
        for (int i = 0; i < 3; ++i) {
            const int ot = wave * 3 + i;          // scalar -> s_cmp/s_cbranch below
            v8f acc = (v8f){};
            acc = wmma_bf16(wa[i][0], b0, acc);
            acc = wmma_bf16(wa[i][1], b1, acc);
            if (ot < 4) {               // q rows (8 consecutive o -> one b128 store)
                union { bf16_t h[8]; uint4 u; } pk;
                #pragma unroll
                for (int r = 0; r < 8; ++r)
                    pk.h[r] = (bf16_t)((acc[r] + bfr[i][r]) * (8.f * LOG2E));
                *(uint4*)(q + (((size_t)b << 12) + n) * 64 + ot * 16 + g * 8) = pk.u;
            } else if (ot < 8) {        // k rows
                union { bf16_t h[8]; uint4 u; } pk;
                #pragma unroll
                for (int r = 0; r < 8; ++r)
                    pk.h[r] = (bf16_t)(acc[r] + bfr[i][r]);
                *(uint4*)(k + (((size_t)b << 12) + n) * 64 + (ot - 4) * 16 + g * 8) = pk.u;
            } else {                    // v rows, stored transposed [c][n]
                #pragma unroll
                for (int r = 0; r < 8; ++r) {
                    const int c = (ot - 8) * 16 + g * 8 + r;
                    vt[(((size_t)b * 64 + c) << 12) + n] = (bf16_t)(acc[r] + bfr[i][r]);
                }
            }
        }
    }
}

// -------------------- kernel 3: flash attention --------------------
// 128 threads = 4 waves, 16 query rows/wave. Double-buffered K/V tiles via
// CDNA5 async global->LDS (ASYNCcnt); WMMA bf16 for S, P@V and the softmax
// row-sum (P @ ones); row-max butterfly on v_permlane16. Output bf16 [N][C].
__global__ __launch_bounds__(128) void k_flash(const bf16_t* __restrict__ q,
                                               const bf16_t* __restrict__ kg,
                                               const bf16_t* __restrict__ vt,
                                               bf16_t* __restrict__ out)
{
    __shared__ bf16_t sK[2][64][72];   // [buf][key][ch] 36-dword rows: conflict-free
    __shared__ bf16_t sV[2][64][72];   // [buf][ch][key]
    __shared__ bf16_t sP[4][16][72];   // per-wave C-layout -> A-layout staging

    const int b    = blockIdx.y;
    const int n0   = blockIdx.x << 6;
    const int tid  = threadIdx.x;
    const int wave = __builtin_amdgcn_readfirstlane(tid >> 5);
    const int lane = tid & 31;
    const int g    = lane >> 4;
    const int ln   = lane & 15;

    const bf16_t* qrow = q + (((size_t)b << 12) + n0 + wave * 16 + ln) * AB_C;
    BFr qa0, qa1;
    qa0.u[0] = *(const uint4*)(qrow +  0 + g * 8);
    qa0.u[1] = *(const uint4*)(qrow + 16 + g * 8);
    qa1.u[0] = *(const uint4*)(qrow + 32 + g * 8);
    qa1.u[1] = *(const uint4*)(qrow + 48 + g * 8);

    BFr ones;
    #pragma unroll
    for (int e = 0; e < 16; ++e) ones.v[e] = (bf16_t)1.0f;

    v8f O[4];
    #pragma unroll
    for (int j = 0; j < 4; ++j) O[j] = (v8f){};
    v8f lvec = (v8f){};
    float m[8];
    #pragma unroll
    for (int r = 0; r < 8; ++r) m[r] = -1e30f;

    const bf16_t* kbase = kg + (((size_t)b << 12)) * AB_C;   // [N][C]
    const bf16_t* vbase = vt + (((size_t)b * AB_C) << 12);   // [C][N]

    const int row  = tid >> 1;
    const int half = tid & 1;

    auto prefetch = [&](int t, int buf) {
        const unsigned long long kp =
            (unsigned long long)(uintptr_t)(kbase + (size_t)(t * 64 + row) * AB_C + half * 32);
        const unsigned long long vp =
            (unsigned long long)(uintptr_t)(vbase + ((size_t)row << 12) + t * 64 + half * 32);
        const unsigned kl = (unsigned)(uintptr_t)&sK[buf][row][half * 32];
        const unsigned vl = (unsigned)(uintptr_t)&sV[buf][row][half * 32];
        asm volatile(
            "global_load_async_to_lds_b128 %0, %2, off\n\t"
            "global_load_async_to_lds_b128 %0, %2, off offset:16\n\t"
            "global_load_async_to_lds_b128 %0, %2, off offset:32\n\t"
            "global_load_async_to_lds_b128 %0, %2, off offset:48\n\t"
            "global_load_async_to_lds_b128 %1, %3, off\n\t"
            "global_load_async_to_lds_b128 %1, %3, off offset:16\n\t"
            "global_load_async_to_lds_b128 %1, %3, off offset:32\n\t"
            "global_load_async_to_lds_b128 %1, %3, off offset:48"
            :: "v"(kl), "v"(vl), "v"(kp), "v"(vp) : "memory");
    };

    prefetch(0, 0);

    for (int t = 0; t < AB_N / 64; ++t) {
        const int cur = t & 1;
        asm volatile("s_wait_asynccnt 0x0" ::: "memory");
        __syncthreads();
        if (t + 1 < AB_N / 64) prefetch(t + 1, cur ^ 1);

        // S = Q @ K^T, four 16x16 tiles
        v8f S[4];
        #pragma unroll
        for (int j = 0; j < 4; ++j) {
            const bf16_t* kr = &sK[cur][j * 16 + ln][0];
            BFr b0, b1;
            b0.u[0] = *(const uint4*)(kr + g * 8);
            b0.u[1] = *(const uint4*)(kr + 16 + g * 8);
            b1.u[0] = *(const uint4*)(kr + 32 + g * 8);
            b1.u[1] = *(const uint4*)(kr + 48 + g * 8);
            v8f acc = (v8f){};
            acc = wmma_bf16(qa0, b0, acc);
            acc = wmma_bf16(qa1, b1, acc);
            S[j] = acc;
        }

        // online softmax (exp2 domain); C/D layout: elem r <-> row g*8+r, col ln
        #pragma unroll
        for (int r = 0; r < 8; ++r) {
            float mx = fmaxf(fmaxf(S[0][r], S[1][r]), fmaxf(S[2][r], S[3][r]));
            mx = fmax_bfly16(mx, 0x67452301u, 0xEFCDAB89u);   // xor 1
            mx = fmax_bfly16(mx, 0x54761032u, 0xDCFE98BAu);   // xor 2
            mx = fmax_bfly16(mx, 0x32107654u, 0xBA98FEDCu);   // xor 4
            mx = fmax_bfly16(mx, 0xFEDCBA98u, 0x76543210u);   // xor 8
            const float mn    = fmaxf(m[r], mx);
            const float alpha = __builtin_amdgcn_exp2f(m[r] - mn);
            m[r] = mn;
            S[0][r] = __builtin_amdgcn_exp2f(S[0][r] - mn);
            S[1][r] = __builtin_amdgcn_exp2f(S[1][r] - mn);
            S[2][r] = __builtin_amdgcn_exp2f(S[2][r] - mn);
            S[3][r] = __builtin_amdgcn_exp2f(S[3][r] - mn);
            lvec[r] *= alpha;
            O[0][r] *= alpha; O[1][r] *= alpha; O[2][r] *= alpha; O[3][r] *= alpha;
        }

        // C-layout -> A-layout via per-wave LDS staging
        #pragma unroll
        for (int j = 0; j < 4; ++j)
            #pragma unroll
            for (int r = 0; r < 8; ++r)
                sP[wave][g * 8 + r][j * 16 + ln] = (bf16_t)S[j][r];

        asm volatile("s_wait_dscnt 0x0" ::: "memory");

        BFr pa0, pa1;
        const bf16_t* prow = &sP[wave][ln][0];
        pa0.u[0] = *(const uint4*)(prow +  0 + g * 8);
        pa0.u[1] = *(const uint4*)(prow + 16 + g * 8);
        pa1.u[0] = *(const uint4*)(prow + 32 + g * 8);
        pa1.u[1] = *(const uint4*)(prow + 48 + g * 8);

        // row-sum on the matrix pipe: l += P @ ones
        lvec = wmma_bf16(pa0, ones, lvec);
        lvec = wmma_bf16(pa1, ones, lvec);

        // O += P @ V
        #pragma unroll
        for (int j = 0; j < 4; ++j) {
            const bf16_t* vr = &sV[cur][j * 16 + ln][0];
            BFr b0, b1;
            b0.u[0] = *(const uint4*)(vr + g * 8);
            b0.u[1] = *(const uint4*)(vr + 16 + g * 8);
            b1.u[0] = *(const uint4*)(vr + 32 + g * 8);
            b1.u[1] = *(const uint4*)(vr + 48 + g * 8);
            v8f acc = O[j];
            acc = wmma_bf16(pa0, b0, acc);
            acc = wmma_bf16(pa1, b1, acc);
            O[j] = acc;
        }
    }

    #pragma unroll
    for (int r = 0; r < 8; ++r) {
        const float inv = __builtin_amdgcn_rcpf(lvec[r]);
        const int rowq = n0 + wave * 16 + g * 8 + r;
        bf16_t* orow = out + (((size_t)b << 12) + rowq) * AB_C;
        #pragma unroll
        for (int j = 0; j < 4; ++j) orow[j * 16 + ln] = (bf16_t)(O[j][r] * inv);
    }
}

// -------------------- kernel 4: output projection (WMMA) + fused residual --------------------
// out[o][n] = W_out @ attn^T + b + xn; B-fragments of attn^T load directly from
// global bf16 [N][C] (contiguous in c) -> no LDS at all.
__global__ __launch_bounds__(128) void k_out(const bf16_t* __restrict__ attnb,
                                             const float* __restrict__ w,
                                             const float* __restrict__ bias,
                                             const float* __restrict__ x,
                                             const float* __restrict__ meanp,
                                             const float* __restrict__ rstdp,
                                             float* __restrict__ out)
{
    const int b    = blockIdx.y;
    const int n0   = blockIdx.x << 6;
    const int tid  = threadIdx.x;
    const int wave = __builtin_amdgcn_readfirstlane(tid >> 5);   // o-tile (scalar)
    const int lane = tid & 31;
    const int g    = lane >> 4;
    const int ln   = lane & 15;

    // A-fragments: W_out rows wave*16+ln (fp32 -> bf16)
    BFr wa0, wa1;
    {
        const float* wr = w + (size_t)(wave * 16 + ln) * 64;
        #pragma unroll
        for (int e = 0; e < 8; ++e) {
            wa0.v[e]     = (bf16_t)wr[g * 8 + e];
            wa0.v[e + 8] = (bf16_t)wr[16 + g * 8 + e];
            wa1.v[e]     = (bf16_t)wr[32 + g * 8 + e];
            wa1.v[e + 8] = (bf16_t)wr[48 + g * 8 + e];
        }
    }
    float bf8[8], mu8[8], rs8[8];
    #pragma unroll
    for (int r = 0; r < 8; ++r) {
        const int o = wave * 16 + g * 8 + r;
        bf8[r] = bias[o];
        mu8[r] = meanp[b * 64 + o];
        rs8[r] = rstdp[b * 64 + o];
    }

    #pragma unroll
    for (int j = 0; j < 4; ++j) {
        const bf16_t* ar = attnb + (((size_t)b << 12) + n0 + j * 16 + ln) * 64;
        BFr b0, b1;
        b0.u[0] = *(const uint4*)(ar + g * 8);
        b0.u[1] = *(const uint4*)(ar + 16 + g * 8);
        b1.u[0] = *(const uint4*)(ar + 32 + g * 8);
        b1.u[1] = *(const uint4*)(ar + 48 + g * 8);
        v8f acc = (v8f){};
        acc = wmma_bf16(wa0, b0, acc);
        acc = wmma_bf16(wa1, b1, acc);
        const int n = n0 + j * 16 + ln;
        #pragma unroll
        for (int r = 0; r < 8; ++r) {
            const int o = wave * 16 + g * 8 + r;
            const size_t oi = (((size_t)b * 64 + o) << 12) + n;
            out[oi] = acc[r] + bf8[r] + (x[oi] - mu8[r]) * rs8[r];
        }
    }
}

// -------------------- launcher --------------------
extern "C" void kernel_launch(void* const* d_in, const int* in_sizes, int n_in,
                              void* d_out, int out_size, void* d_ws, size_t ws_size,
                              hipStream_t stream)
{
    const float* x     = (const float*)d_in[0];
    const float* w_qkv = (const float*)d_in[1];
    const float* b_qkv = (const float*)d_in[2];
    const float* w_out = (const float*)d_in[3];
    const float* b_out = (const float*)d_in[4];
    float* out = (float*)d_out;

    char* ws = (char*)d_ws;
    float*  meanp = (float*)ws;                    // 1 KB [B*C]
    float*  rstdp = meanp + AB_B * AB_C;           // 1 KB [B*C]
    bf16_t* q     = (bf16_t*)(ws + (4u  << 20));   // 2 MB [B][N][C] bf16 (pre-scaled)
    bf16_t* k     = (bf16_t*)(ws + (6u  << 20));   // 2 MB [B][N][C] bf16
    bf16_t* vt    = (bf16_t*)(ws + (8u  << 20));   // 2 MB [B][C][N] bf16
    bf16_t* attnb = (bf16_t*)(ws + (10u << 20));   // 2 MB [B][N][C] bf16

    k_stats<<<dim3(AB_B * AB_C), dim3(256), 0, stream>>>(x, meanp, rstdp);
    k_qkv  <<<dim3(AB_N / 64, AB_B), dim3(128), 0, stream>>>(x, meanp, rstdp, w_qkv, b_qkv, q, k, vt);
    k_flash<<<dim3(AB_N / 64, AB_B), dim3(128), 0, stream>>>(q, k, vt, attnb);
    k_out  <<<dim3(AB_N / 64, AB_B), dim3(128), 0, stream>>>(attnb, w_out, b_out, x, meanp, rstdp, out);
}
